// MinGPT_71897752535323
// MI455X (gfx1250) — compile-verified, matrix operand order checked by reading, not run on hardware
//
#include <hip/hip_runtime.h>
#include <hip/hip_bf16.h>
#include <math.h>

// Problem dims (compile-time, from reference)
constexpr int CB = 4, CS = 1024, CE = 1024, CH = 16, CL = 12, CV = 1024, CD = 64;
constexpr int CM = CB * CS;          // 4096 rows of activations

typedef __bf16 bf16_t;
typedef __attribute__((ext_vector_type(16))) __bf16 v16bf;
typedef __attribute__((ext_vector_type(8)))  __bf16 v8bf;
typedef __attribute__((ext_vector_type(8)))  float  v8f;

// 16-bit A/B fragment K-index mapping (ISA 7.12.2, 16-bit A 16x32 table):
// VGPR i, lane-half lh: k0 = (i<4?0:16) + lh*8 + (i%4)*2 ; k1 = k0+1
// => per lane the 16 values are two contiguous 8-element runs:
//    k in [lh*8, lh*8+8) and [16+lh*8, 16+lh*8+8)
__device__ __forceinline__ int fragk(int i, int lh) {
    return ((i >> 2) << 4) + lh * 8 + ((i & 3) << 1);
}

__device__ __forceinline__ v16bf join8(v8bf lo, v8bf hi) {
    return __builtin_shufflevector(lo, hi, 0, 1, 2, 3, 4, 5, 6, 7,
                                   8, 9, 10, 11, 12, 13, 14, 15);
}

// load a 16-bit fragment as two contiguous 16B runs (base = row start, runs at
// lh*8 and 16+lh*8)
__device__ __forceinline__ v16bf load_frag_contig(const bf16_t* base, int lh) {
    v8bf lo = *(const v8bf*)(base + lh * 8);
    v8bf hi = *(const v8bf*)(base + 16 + lh * 8);
    return join8(lo, hi);
}

__device__ __forceinline__ v8f wmma_bf16(v16bf a, v16bf b, v8f c) {
    return __builtin_amdgcn_wmma_f32_16x16x32_bf16(false, a, false, b, (short)0, c, false, false);
}

__device__ __forceinline__ float gelu_exact(float x) {
    return 0.5f * x * (1.0f + erff(x * 0.70710678118654752f));
}

// ---- CDNA5 async global->LDS (ASYNCcnt-tracked), inline asm per VGLOBAL encoding
__device__ __forceinline__ void async_copy_b128(void* lds_dst, const void* gsrc) {
    unsigned lds = (unsigned)(unsigned long long)lds_dst;   // low 32 bits = LDS byte addr
    unsigned long long ga = (unsigned long long)gsrc;
    asm volatile("global_load_async_to_lds_b128 %0, %1, off"
                 :: "v"(lds), "v"(ga) : "memory");
}
__device__ __forceinline__ void wait_async_le3() {
    asm volatile("s_wait_asynccnt 0x3" ::: "memory");
}
__device__ __forceinline__ void wait_async_le0() {
    asm volatile("s_wait_asynccnt 0x0" ::: "memory");
}

// ---------------------------------------------------------------- embed
__global__ void embed_kernel(const int* __restrict__ tokens,
                             const float* __restrict__ tok_emb,
                             const float* __restrict__ pos_emb,
                             float* __restrict__ h) {
    size_t idx = (size_t)blockIdx.x * blockDim.x + threadIdx.x;
    if (idx >= (size_t)CM * CE) return;
    int e  = (int)(idx % CE);
    int bs = (int)(idx / CE);
    int s  = bs % CS;
    int t  = tokens[bs];
    h[idx] = tok_emb[(size_t)t * CE + e] + pos_emb[(size_t)s * CE + e];
}

// ---------------------------------------------------------------- f32 -> bf16
__global__ void f32_to_bf16_kernel(const float* __restrict__ in,
                                   bf16_t* __restrict__ out, size_t n) {
    size_t idx = (size_t)blockIdx.x * blockDim.x + threadIdx.x;
    if (idx < n) out[idx] = (bf16_t)in[idx];
}

// ---------------------------------------------------------------- LayerNorm -> bf16
__global__ __launch_bounds__(256)
void ln_bf16_kernel(const float* __restrict__ x, const float* __restrict__ w,
                    const float* __restrict__ b, bf16_t* __restrict__ out) {
    const int row = blockIdx.x;
    const int tid = threadIdx.x;
    const float* xr = x + (size_t)row * CE;
    __shared__ float red[256];

    float v[4];
    float s = 0.0f;
#pragma unroll
    for (int i = 0; i < 4; ++i) { v[i] = xr[tid + i * 256]; s += v[i]; }
    red[tid] = s; __syncthreads();
    for (int st = 128; st > 0; st >>= 1) {
        if (tid < st) red[tid] += red[tid + st];
        __syncthreads();
    }
    const float mean = red[0] * (1.0f / CE);
    __syncthreads();

    float s2 = 0.0f;
#pragma unroll
    for (int i = 0; i < 4; ++i) { float d = v[i] - mean; s2 += d * d; }
    red[tid] = s2; __syncthreads();
    for (int st = 128; st > 0; st >>= 1) {
        if (tid < st) red[tid] += red[tid + st];
        __syncthreads();
    }
    const float rstd = rsqrtf(red[0] * (1.0f / CE) + 1e-5f);

#pragma unroll
    for (int i = 0; i < 4; ++i) {
        int c = tid + i * 256;
        out[(size_t)row * CE + c] = (bf16_t)((v[i] - mean) * rstd * w[c] + b[c]);
    }
}

// ---------------------------------------------------------------- qkv repack
// qkv f32 [b,s, 3, h, d]  ->  Qbf/Kbf bf16 [b,h,s,d] (Q pre-scaled by 1/sqrt(D)),
//                             Vt bf16 [b,h,d,s] (transposed for P@V B-fragments)
__global__ void qkv_repack_kernel(const float* __restrict__ qkv,
                                  bf16_t* __restrict__ Qb, bf16_t* __restrict__ Kb,
                                  bf16_t* __restrict__ Vt) {
    size_t idx = (size_t)blockIdx.x * blockDim.x + threadIdx.x;
    if (idx >= (size_t)CM * CE) return;
    const int d  = (int)(idx % CD);
    const int hh = (int)((idx / CD) % CH);
    const int bs = (int)(idx / CE);
    const int s  = bs % CS;
    const int b  = bs / CS;

    const float* base = qkv + (size_t)bs * 3 * CE + hh * CD + d;
    const float q = base[0];
    const float k = base[CE];
    const float v = base[2 * CE];

    const size_t bh = (size_t)b * CH + hh;
    Qb[(bh * CS + s) * CD + d] = (bf16_t)(q * 0.125f);
    Kb[(bh * CS + s) * CD + d] = (bf16_t)k;
    Vt[(bh * CD + d) * CS + s] = (bf16_t)v;
}

// ---------------------------------------------------------------- tiled bf16 GEMM
// C[M,N] = A[M,K](bf16) @ Bw[K,N](bf16)
// EPI: 0 = f32 out, 1 = gelu -> bf16 out. HAS_BIAS/HAS_RES compile-time.
// Double-buffered LDS staged with global_load_async_to_lds_b128 (3 async ops
// per wave per stage), pipelined with s_wait_asynccnt.
constexpr int BM = 128, BN = 64, BK = 32;

template <int EPI, bool HAS_BIAS, bool HAS_RES>
__global__ __launch_bounds__(256)
void gemm_bf16_kernel(const bf16_t* __restrict__ A, const bf16_t* __restrict__ Bw,
                      const float* __restrict__ bias, const float* __restrict__ residual,
                      float* __restrict__ outF, bf16_t* __restrict__ outB,
                      int M, int N, int K) {
    __shared__ bf16_t As[2][BM][BK + 8];   // row stride 80B (16B aligned)
    __shared__ bf16_t Bs[2][BK][BN + 8];   // row stride 144B (16B aligned)

    const int m0 = blockIdx.y * BM;
    const int n0 = blockIdx.x * BN;
    const int tid  = threadIdx.x;
    const int wave = tid >> 5;
    const int lane = tid & 31;
    const int waveM = wave >> 1;     // 0..3  -> 32-row slab
    const int waveN = wave & 1;      // 0..1  -> 32-col slab
    const int lh = lane >> 4;        // lane half
    const int ln = lane & 15;

    // per-thread staging coordinates
    const int ar = tid >> 1, ac = (tid & 1) * 16;      // A: 16 elems (2x b128)
    const int br = tid >> 3, bc = (tid & 7) * 8;       // B: 8 elems  (1x b128)

    auto stage = [&](int k0, int buf) {
        const bf16_t* asrc = A + (size_t)(m0 + ar) * K + (k0 + ac);
        async_copy_b128(&As[buf][ar][ac],     asrc);
        async_copy_b128(&As[buf][ar][ac + 8], asrc + 8);
        const bf16_t* bsrc = Bw + (size_t)(k0 + br) * N + (n0 + bc);
        async_copy_b128(&Bs[buf][br][bc], bsrc);
    };

    v8f acc[2][2] = {};
    const int nk = K / BK;

    stage(0, 0);
    for (int i = 0; i < nk; ++i) {
        const int buf = i & 1;
        if (i + 1 < nk) { stage((i + 1) * BK, buf ^ 1); wait_async_le3(); }
        else            { wait_async_le0(); }
        __syncthreads();

        v16bf afrag[2], bfrag[2];
#pragma unroll
        for (int ms = 0; ms < 2; ++ms) {
            const int row = waveM * 32 + ms * 16 + ln;
            afrag[ms] = load_frag_contig(&As[buf][row][0], lh);
        }
#pragma unroll
        for (int ns = 0; ns < 2; ++ns) {
            const int col = waveN * 32 + ns * 16 + ln;
#pragma unroll
            for (int j = 0; j < 8; ++j) {
                const int kk = fragk(j, lh);
                bfrag[ns][2 * j]     = Bs[buf][kk][col];
                bfrag[ns][2 * j + 1] = Bs[buf][kk + 1][col];
            }
        }
#pragma unroll
        for (int ms = 0; ms < 2; ++ms)
#pragma unroll
            for (int ns = 0; ns < 2; ++ns)
                acc[ms][ns] = wmma_bf16(afrag[ms], bfrag[ns], acc[ms][ns]);
        __syncthreads();
    }

    // epilogue: C layout — VGPR r: lanes0-15 row=r, lanes16-31 row=r+8; col = ln
#pragma unroll
    for (int ms = 0; ms < 2; ++ms)
#pragma unroll
        for (int ns = 0; ns < 2; ++ns)
#pragma unroll
            for (int r = 0; r < 8; ++r) {
                const int row = m0 + waveM * 32 + ms * 16 + r + lh * 8;
                const int col = n0 + waveN * 32 + ns * 16 + ln;
                float v = acc[ms][ns][r];
                if constexpr (HAS_BIAS) v += bias[col];
                const size_t idx = (size_t)row * N + col;
                if constexpr (EPI == 1) {
                    outB[idx] = (bf16_t)gelu_exact(v);
                } else {
                    if constexpr (HAS_RES) v += residual[idx];
                    outF[idx] = v;
                }
            }
}

// ---------------------------------------------------------------- flash attention
// One wave per (b, head, 16-query block). Streams keys in 32-wide tiles from the
// repacked bf16 buffers; every fragment load is a contiguous 16B run.
__global__ __launch_bounds__(32)
void attn_kernel(const bf16_t* __restrict__ Qb, const bf16_t* __restrict__ Kb,
                 const bf16_t* __restrict__ Vt, float* __restrict__ h) {
    const int q0 = blockIdx.x * 16;
    const int hh = blockIdx.y;
    const int b  = blockIdx.z;
    const int lane = threadIdx.x;
    const int lh = lane >> 4;
    const int ln = lane & 15;

    const size_t bh = (size_t)b * CH + hh;
    const bf16_t* Qh = Qb + bh * CS * CD;   // [s][d]
    const bf16_t* Kh = Kb + bh * CS * CD;   // [s][d]
    const bf16_t* Vh = Vt + bh * CD * CS;   // [d][s]

    // Q fragments (16x64 -> two 16x32 A-frags), already scaled by 1/sqrt(D)
    v16bf qf[2];
#pragma unroll
    for (int dstep = 0; dstep < 2; ++dstep)
        qf[dstep] = load_frag_contig(Qh + (size_t)(q0 + ln) * CD + dstep * 32, lh);

    float mrow[8], lrow[8];
    v8f o[4] = {};
#pragma unroll
    for (int r = 0; r < 8; ++r) { mrow[r] = -1e30f; lrow[r] = 0.0f; }

    __shared__ bf16_t Pb[16][40];   // row stride 80B (16B aligned)

    for (int kt = 0; kt <= q0 + 15; kt += 32) {
        // scores: two 16x16 tiles over keys [kt, kt+31]
        v8f sc[2] = {};
#pragma unroll
        for (int nt = 0; nt < 2; ++nt) {
            const int key = kt + nt * 16 + ln;
#pragma unroll
            for (int dstep = 0; dstep < 2; ++dstep) {
                v16bf kf = load_frag_contig(Kh + (size_t)key * CD + dstep * 32, lh);
                sc[nt] = wmma_bf16(qf[dstep], kf, sc[nt]);
            }
        }

        // causal mask + online softmax, per row
        float alpha[8];
#pragma unroll
        for (int r = 0; r < 8; ++r) {
            const int row = q0 + r + lh * 8;
            float s0 = sc[0][r];
            float s1 = sc[1][r];
            if (kt + ln > row)      s0 = -1e30f;
            if (kt + 16 + ln > row) s1 = -1e30f;
            float mx = fmaxf(s0, s1);
#pragma unroll
            for (int off = 8; off >= 1; off >>= 1)
                mx = fmaxf(mx, __shfl_xor(mx, off, 32));
            const float mnew = fmaxf(mrow[r], mx);
            const float a  = __expf(mrow[r] - mnew);
            const float p0 = __expf(s0 - mnew);
            const float p1 = __expf(s1 - mnew);
            float ps = p0 + p1;
#pragma unroll
            for (int off = 8; off >= 1; off >>= 1)
                ps += __shfl_xor(ps, off, 32);
            lrow[r] = lrow[r] * a + ps;
            mrow[r] = mnew;
            alpha[r] = a;
            Pb[r + lh * 8][ln]      = (bf16_t)p0;
            Pb[r + lh * 8][16 + ln] = (bf16_t)p1;
        }
        __syncthreads();

        // rescale running output
#pragma unroll
        for (int t = 0; t < 4; ++t)
#pragma unroll
            for (int r = 0; r < 8; ++r) o[t][r] *= alpha[r];

        // P as A-fragment (16x32 over keys): two contiguous 16B LDS runs
        v16bf pf = load_frag_contig(&Pb[ln][0], lh);

        // P @ V : 4 d-tiles of 16; Vt rows are contiguous along keys
#pragma unroll
        for (int t = 0; t < 4; ++t) {
            v16bf vf = load_frag_contig(Vh + (size_t)(t * 16 + ln) * CS + kt, lh);
            o[t] = wmma_bf16(pf, vf, o[t]);
        }
        __syncthreads();
    }

    // h += O / l   (residual add, unique elements per workgroup)
#pragma unroll
    for (int t = 0; t < 4; ++t)
#pragma unroll
        for (int r = 0; r < 8; ++r) {
            const int row = q0 + r + lh * 8;
            const int d   = t * 16 + ln;
            const size_t idx = ((size_t)b * CS + row) * CE + hh * CD + d;
            h[idx] += o[t][r] / lrow[r];
        }
}

// ---------------------------------------------------------------- host
extern "C" void kernel_launch(void* const* d_in, const int* in_sizes, int n_in,
                              void* d_out, int out_size, void* d_ws, size_t ws_size,
                              hipStream_t stream) {
    const int*   tokens  = (const int*)d_in[0];
    const float* tok_emb = (const float*)d_in[1];
    const float* pos_emb = (const float*)d_in[2];
    const float* ln_w    = (const float*)d_in[3];
    const float* ln_b    = (const float*)d_in[4];
    const float* qkv_w   = (const float*)d_in[5];
    const float* qkv_b   = (const float*)d_in[6];
    const float* fc1_w   = (const float*)d_in[7];
    const float* fc1_b   = (const float*)d_in[8];
    const float* fc2_w   = (const float*)d_in[9];
    const float* fc2_b   = (const float*)d_in[10];
    const float* lnf_w   = (const float*)d_in[11];
    const float* lnf_b   = (const float*)d_in[12];
    const float* head_w  = (const float*)d_in[13];
    float* logits = (float*)d_out;

    char* ws = (char*)d_ws;
    size_t off = 0;
    auto alloc = [&](size_t bytes) -> char* {
        char* p = ws + off;
        off += (bytes + 255) & ~(size_t)255;
        return p;
    };

    const size_t nWqkv = (size_t)CL * CE * 3 * CE;
    const size_t nWf1  = (size_t)CL * CE * 4 * CE;
    const size_t nWf2  = (size_t)CL * 4 * CE * CE;
    const size_t nWhd  = (size_t)CE * CV;
    const size_t nBHSD = (size_t)CB * CH * CS * CD;

    bf16_t* wqkv = (bf16_t*)alloc(nWqkv * sizeof(bf16_t));
    bf16_t* wf1  = (bf16_t*)alloc(nWf1  * sizeof(bf16_t));
    bf16_t* wf2  = (bf16_t*)alloc(nWf2  * sizeof(bf16_t));
    bf16_t* whd  = (bf16_t*)alloc(nWhd  * sizeof(bf16_t));
    float*  h    = (float*) alloc((size_t)CM * CE * sizeof(float));
    bf16_t* xln  = (bf16_t*)alloc((size_t)CM * CE * sizeof(bf16_t));
    float*  qkv  = (float*) alloc((size_t)CM * 3 * CE * sizeof(float));
    bf16_t* y    = (bf16_t*)alloc((size_t)CM * 4 * CE * sizeof(bf16_t));
    bf16_t* Qbf  = (bf16_t*)alloc(nBHSD * sizeof(bf16_t));
    bf16_t* Kbf  = (bf16_t*)alloc(nBHSD * sizeof(bf16_t));
    bf16_t* Vtb  = (bf16_t*)alloc(nBHSD * sizeof(bf16_t));

    auto cvt = [&](const float* src, bf16_t* dst, size_t n) {
        f32_to_bf16_kernel<<<dim3((unsigned)((n + 255) / 256)), 256, 0, stream>>>(src, dst, n);
    };
    cvt(qkv_w, wqkv, nWqkv);
    cvt(fc1_w, wf1, nWf1);
    cvt(fc2_w, wf2, nWf2);
    cvt(head_w, whd, nWhd);

    // embed
    {
        size_t n = (size_t)CM * CE;
        embed_kernel<<<dim3((unsigned)((n + 255) / 256)), 256, 0, stream>>>(tokens, tok_emb, pos_emb, h);
    }

    const unsigned nElemBlocks = (unsigned)(((size_t)CM * CE + 255) / 256);

    for (int l = 0; l < CL; ++l) {
        const float* lw = ln_w + (size_t)l * CE;
        const float* lb = ln_b + (size_t)l * CE;

        // x = LN(h)
        ln_bf16_kernel<<<CM, 256, 0, stream>>>(h, lw, lb, xln);

        // qkv = x @ Wqkv + b
        gemm_bf16_kernel<0, true, false><<<dim3(3 * CE / BN, CM / BM), 256, 0, stream>>>(
            xln, wqkv + (size_t)l * CE * 3 * CE, qkv_b + (size_t)l * 3 * CE,
            nullptr, qkv, nullptr, CM, 3 * CE, CE);

        // repack to WMMA-friendly bf16 layouts (Q scaled by 1/sqrt(D))
        qkv_repack_kernel<<<nElemBlocks, 256, 0, stream>>>(qkv, Qbf, Kbf, Vtb);

        // h += attention(Q,K,V)
        attn_kernel<<<dim3(CS / 16, CH, CB), 32, 0, stream>>>(Qbf, Kbf, Vtb, h);

        // x2 = LN(h)  (same LN params, as in source)
        ln_bf16_kernel<<<CM, 256, 0, stream>>>(h, lw, lb, xln);

        // y = gelu(x2 @ W1 + b1)  -> bf16
        gemm_bf16_kernel<1, true, false><<<dim3(4 * CE / BN, CM / BM), 256, 0, stream>>>(
            xln, wf1 + (size_t)l * CE * 4 * CE, fc1_b + (size_t)l * 4 * CE,
            nullptr, nullptr, y, CM, 4 * CE, CE);

        // h = h + y @ W2 + b2
        gemm_bf16_kernel<0, true, true><<<dim3(CE / BN, CM / BM), 256, 0, stream>>>(
            y, wf2 + (size_t)l * 4 * CE * CE, fc2_b + (size_t)l * CE,
            h, h, nullptr, CM, CE, 4 * CE);
    }

    // final LN + head
    ln_bf16_kernel<<<CM, 256, 0, stream>>>(h, lnf_w, lnf_b, xln);
    gemm_bf16_kernel<0, false, false><<<dim3(CV / BN, CM / BM), 256, 0, stream>>>(
        xln, whd, nullptr, nullptr, logits, nullptr, CM, CV, CE);
}
